// NeuralODESolver_63737314672942
// MI455X (gfx1250) — matrix operand order repaired and checked by reading, not compiled, not run on hardware
//
#include <hip/hip_runtime.h>

#define DT_STEP 0.1f
#define NB 65536
#define DD 64
#define HH 128

typedef _Float16 v16h __attribute__((ext_vector_type(16)));
typedef float    v8f  __attribute__((ext_vector_type(8)));

union U16h { v16h v; uint4 q[2]; unsigned u[8]; };

__device__ __forceinline__ unsigned pkh(float a, float b) {
  typedef __fp16 hv2 __attribute__((ext_vector_type(2)));
  hv2 r = __builtin_amdgcn_cvt_pkrtz(a, b);    // [15:0]=a, [31:16]=b
  return __builtin_bit_cast(unsigned, r);
}

__device__ __forceinline__ float act_tanh(float x) {
#if __has_builtin(__builtin_amdgcn_tanhf)
  return __builtin_amdgcn_tanhf(x);            // v_tanh_f32 (gfx1250)
#elif __has_builtin(__builtin_amdgcn_tanh_f32)
  return __builtin_amdgcn_tanh_f32(x);
#else
  return tanhf(x);
#endif
}

// Swap data between lanes L and L^16 (cross-half) via v_permlanex16_b32:
// pure VALU, co-executes with XDL WMMA (vs ds_bpermute burning the LDS pipe).
__device__ __forceinline__ unsigned xswap16(unsigned x) {
#if __has_builtin(__builtin_amdgcn_permlanex16)
  return (unsigned)__builtin_amdgcn_permlanex16((int)x, (int)x, 0x76543210,
                                                0xfedcba98, true, false);
#else
  return (unsigned)__shfl_xor((int)x, 16, 32);
#endif
}

// Scheduler fence: keep A-operand loads near their consuming WMMA.
__device__ __forceinline__ void sbar() {
#if __has_builtin(__builtin_amdgcn_sched_barrier)
  __builtin_amdgcn_sched_barrier(0);
#endif
}

// acc[r] <- b[m0 + 8*h16 + r]  (two broadcast b128 loads)
__device__ __forceinline__ v8f bias_init(const float* b, int m0, int h16) {
  const float4* p = (const float4*)(b + m0 + 8 * h16);
  float4 x = p[0], y = p[1];
  v8f r;
  r[0]=x.x; r[1]=x.y; r[2]=x.z; r[3]=x.w;
  r[4]=y.x; r[5]=y.y; r[6]=y.z; r[7]=y.w;
  return r;
}

// Build B operand (32x16 f16) for K-chunk from two C-layout f32 tiles.
// B vgpr v: lanes<16 hold K=2v,2v+1 (tile0 M=2v..), lanes>=16 hold K=16+2v (tile1).
__device__ __forceinline__ void pack_pair(const v8f t0, const v8f t1, int h16, unsigned* outc) {
#pragma unroll
  for (int v = 0; v < 4; ++v) {
    unsigned p0 = pkh(t0[2*v], t0[2*v+1]);
    unsigned p1 = pkh(t1[2*v], t1[2*v+1]);
    unsigned s0 = xswap16(p0);
    unsigned s1 = xswap16(p1);
    outc[v]     = h16 ? s1 : p0;
    outc[v + 4] = h16 ? p1 : s0;
  }
}

// One 16x16 output tile: A = weight rows [m0..m0+15] x K (f16, row-major, stride K),
// B = batch chunks, accumulate over KC chunks of 32.
template <int KC>
__device__ __forceinline__ v8f gemm_tiles(const _Float16* __restrict__ W, int m0,
                                          int col, int h16,
                                          const unsigned (*Bch)[8], v8f acc) {
  const int K = KC * 32;
  const _Float16* wrow = W + (size_t)(m0 + col) * K;
#pragma unroll
  for (int kc = 0; kc < KC; ++kc) {
    U16h A;
    const _Float16* p = wrow + kc * 32 + 16 * h16;  // A layout: lanes<16 K=0..15, >=16 K=16..31
    A.q[0] = *(const uint4*)(p);
    A.q[1] = *(const uint4*)(p + 8);                // K=8..15 / 24..31
    U16h Bv;
#pragma unroll
    for (int v = 0; v < 8; ++v) Bv.u[v] = Bch[kc][v];
    acc = __builtin_amdgcn_wmma_f32_16x16x32_f16(false, A.v, false, Bv.v,
                                                 (short)0, acc, false, false);
  }
  return acc;
}

__global__ void k_init(unsigned* slot) { *slot = 0u; }

__global__ void k_redmax(const float* __restrict__ td, unsigned* slot, int n) {
  float m = 0.f;
  for (int i = blockIdx.x * blockDim.x + threadIdx.x; i < n; i += gridDim.x * blockDim.x)
    m = fmaxf(m, fabsf(td[i]));
#pragma unroll
  for (int o = 16; o > 0; o >>= 1) m = fmaxf(m, __shfl_xor(m, o, 32));
  if ((threadIdx.x & 31) == 0) atomicMax(slot, __float_as_uint(m));  // floats >=0: bit-monotonic
}

// Transpose weights to f16, [out][in] row-major (A-operand friendly).
__global__ void k_prep(const float* __restrict__ W1, const float* __restrict__ W2,
                       const float* __restrict__ W3, _Float16* __restrict__ w1t,
                       _Float16* __restrict__ w2t, _Float16* __restrict__ w3t) {
  int i = blockIdx.x * 256 + threadIdx.x;
  if (i < 8192) {                       // W1z: (64,128) -> w1t[128][64]
    int m = i >> 6, k = i & 63;
    w1t[m * 64 + k] = (_Float16)W1[k * 128 + m];
  } else if (i < 24576) {               // W2: (128,128) -> w2t[128][128]
    int j = i - 8192, m = j >> 7, k = j & 127;
    w2t[m * 128 + k] = (_Float16)W2[k * 128 + m];
  } else if (i < 32768) {               // W3: (128,64) -> w3t[64][128]
    int j = i - 24576, m = j >> 7, k = j & 127;
    w3t[m * 128 + k] = (_Float16)W3[k * 64 + m];
  }
}

__global__ __launch_bounds__(256, 1) void ode_kernel(
    const float* __restrict__ z, const float* __restrict__ td,
    const float* __restrict__ b1, const float* __restrict__ wtv,
    const float* __restrict__ b2, const float* __restrict__ b3,
    const _Float16* __restrict__ w1t, const _Float16* __restrict__ w2t,
    const _Float16* __restrict__ w3t, const unsigned* __restrict__ maxslot,
    float* __restrict__ out) {
  __shared__ float sb1[HH], swt[HH], sb2[HH], sb3[DD];
  int tid = threadIdx.x;
  if (tid < HH) { sb1[tid] = b1[tid]; swt[tid] = wtv[tid]; sb2[tid] = b2[tid]; }
  if (tid < DD) { sb3[tid] = b3[tid]; }
  __syncthreads();

  const int lane = tid & 31, h16 = lane >> 4, col = lane & 15;
  const int base = (blockIdx.x * 8 + (tid >> 5)) * 16;
  const int row = base + col;

  const float mx = __uint_as_float(*maxslot);
  const int steps = (int)ceilf(mx / DT_STEP);
  const float dtl = (steps > 0) ? td[row] / (float)steps : 0.f;

  // z state, transposed C-layout: tile t, reg r, lane -> (d = 16t + 8*h16 + r, batch = col)
  v8f zs[4];
  const float* zr = z + (size_t)row * DD;
#pragma unroll
  for (int t = 0; t < 4; ++t) {
    float4 a = *(const float4*)(zr + t * 16 + 8 * h16);
    float4 b = *(const float4*)(zr + t * 16 + 8 * h16 + 4);
    zs[t][0]=a.x; zs[t][1]=a.y; zs[t][2]=a.z; zs[t][3]=a.w;
    zs[t][4]=b.x; zs[t][5]=b.y; zs[t][6]=b.z; zs[t][7]=b.w;
  }

  // Laundered zero offset: loop-variant as far as the compiler knows, so LICM
  // cannot hoist the loop-invariant weight loads out of the step loop (which
  // spilled ~512 VGPRs to scratch). Adding an int to the kernarg pointers
  // keeps address-space inference intact -> global_load (saddr), not flat.
  int lz = 0;

  for (int s = 0; s < steps; ++s) {
    const float tt = s * DT_STEP;

    asm volatile("" : "+s"(lz));
    const _Float16* w1p = w1t + lz;
    const _Float16* w2p = w2t + lz;
    const _Float16* w3p = w3t + lz;

    unsigned zB[2][8];
    pack_pair(zs[0], zs[1], h16, zB[0]);
    pack_pair(zs[2], zs[3], h16, zB[1]);
    sbar();

    // Layer 1: H1^T = tanh(W1z^T (128x64) @ Z^T + b1 + t*Wt)
    unsigned h1B[4][8];
#pragma unroll
    for (int c = 0; c < 4; ++c) {
      v8f a0 = bias_init(sb1, (2*c) * 16, h16);
      v8f w0 = bias_init(swt, (2*c) * 16, h16);
      v8f a1v = bias_init(sb1, (2*c+1) * 16, h16);
      v8f w1v = bias_init(swt, (2*c+1) * 16, h16);
#pragma unroll
      for (int r = 0; r < 8; ++r) { a0[r] += tt * w0[r]; a1v[r] += tt * w1v[r]; }
      a0  = gemm_tiles<2>(w1p, (2*c) * 16, col, h16, zB, a0);
      a1v = gemm_tiles<2>(w1p, (2*c+1) * 16, col, h16, zB, a1v);
#pragma unroll
      for (int r = 0; r < 8; ++r) { a0[r] = act_tanh(a0[r]); a1v[r] = act_tanh(a1v[r]); }
      pack_pair(a0, a1v, h16, h1B[c]);
      sbar();
    }

    // Layer 2: H2^T = tanh(W2^T (128x128) @ H1^T + b2)
    unsigned h2B[4][8];
#pragma unroll
    for (int c = 0; c < 4; ++c) {
      v8f a0 = bias_init(sb2, (2*c) * 16, h16);
      v8f a1v = bias_init(sb2, (2*c+1) * 16, h16);
      a0  = gemm_tiles<4>(w2p, (2*c) * 16, col, h16, h1B, a0);
      a1v = gemm_tiles<4>(w2p, (2*c+1) * 16, col, h16, h1B, a1v);
#pragma unroll
      for (int r = 0; r < 8; ++r) { a0[r] = act_tanh(a0[r]); a1v[r] = act_tanh(a1v[r]); }
      pack_pair(a0, a1v, h16, h2B[c]);
      sbar();
    }

    // Layer 3: dZ^T = W3^T (64x128) @ H2^T + b3 ;  z += dz * dt_row
#pragma unroll
    for (int t = 0; t < 4; ++t) {
      v8f acc = bias_init(sb3, t * 16, h16);
      acc = gemm_tiles<4>(w3p, t * 16, col, h16, h2B, acc);
#pragma unroll
      for (int r = 0; r < 8; ++r) zs[t][r] += acc[r] * dtl;
      sbar();
    }
  }

  float* orow = out + (size_t)row * DD;
#pragma unroll
  for (int t = 0; t < 4; ++t) {
    float4 a, b;
    a.x=zs[t][0]; a.y=zs[t][1]; a.z=zs[t][2]; a.w=zs[t][3];
    b.x=zs[t][4]; b.y=zs[t][5]; b.z=zs[t][6]; b.w=zs[t][7];
    *(float4*)(orow + t * 16 + 8 * h16) = a;
    *(float4*)(orow + t * 16 + 8 * h16 + 4) = b;
  }
}

extern "C" void kernel_launch(void* const* d_in, const int* in_sizes, int n_in,
                              void* d_out, int out_size, void* d_ws, size_t ws_size,
                              hipStream_t stream) {
  const float* z  = (const float*)d_in[0];
  const float* td = (const float*)d_in[1];
  const float* W1 = (const float*)d_in[2];
  const float* b1 = (const float*)d_in[3];
  const float* W2 = (const float*)d_in[4];
  const float* b2 = (const float*)d_in[5];
  const float* W3 = (const float*)d_in[6];
  const float* b3 = (const float*)d_in[7];
  float* out = (float*)d_out;

  unsigned char* ws = (unsigned char*)d_ws;
  unsigned* slot = (unsigned*)ws;
  _Float16* w1t = (_Float16*)(ws + 64);
  _Float16* w2t = (_Float16*)(ws + 64 + 16384);
  _Float16* w3t = (_Float16*)(ws + 64 + 16384 + 32768);

  hipLaunchKernelGGL(k_init, dim3(1), dim3(1), 0, stream, slot);
  hipLaunchKernelGGL(k_redmax, dim3(64), dim3(256), 0, stream, td, slot, NB);
  hipLaunchKernelGGL(k_prep, dim3(128), dim3(256), 0, stream, W1, W2, W3, w1t, w2t, w3t);
  hipLaunchKernelGGL(ode_kernel, dim3(NB / 128), dim3(256), 0, stream,
                     z, td, b1, W1 + 64 * HH, b2, b3, w1t, w2t, w3t, slot, out);
}
